// EncoderRNN_20375324852342
// MI455X (gfx1250) — matrix operand order repaired and checked by reading, not compiled, not run on hardware
//
#include <hip/hip_runtime.h>

// ---------------------------------------------------------------------------
// EncoderRNN (stacked LSTM sharing one (h,c) across layers+time) for MI455X.
//
//   (1) prep:       bias = b_ih+b_hh; zero h double-buffer, final-c, barrier
//   (2) cvt_bf16:   x, W_ih, W_hh -> bf16 (RNE)
//   (3) xproj_wmma: Gx[l][t][4H] = Wih_bf16[l] @ x_bf16[t] + bias[l]
//                   v_wmma_f32_16x16x32_bf16, K-loop register double-buffered
//   (4) lstm_recur: persistent 64-WG kernel, 16384 sequential cell steps.
//                   Per-step 128KB weight tile staged into LDS by the Tensor
//                   Data Mover (TENSOR_LOAD_TO_LDS, double-buffered,
//                   s_wait_tensorcnt) -> weight fetch overlaps compute+sync.
//   (5) head:       out = W_out @ c_final + b_out
// ---------------------------------------------------------------------------

typedef __attribute__((ext_vector_type(16))) __bf16 bf16x16;
typedef __attribute__((ext_vector_type(8)))  float  f32x8;

#define H_      1024
#define IN_     1024
#define L_      8
#define T_      2048
#define G4H     4096            // 4*H
#define NBLK_R  64              // persistent blocks in recurrence
#define NSTEP   (T_ * L_)       // 16384 sequential cell steps
#define WBYTES  (64 * H_ * 2)   // one LDS weight tile: 64 rows x 1024 bf16 = 128KB
#define DYN_LDS (2 * WBYTES + (H_ + 64 + 16) * 4)   // 2 tiles + hlds + glds + clds

#if defined(__HIP_DEVICE_COMPILE__) && __has_builtin(__builtin_amdgcn_tensor_load_to_lds)
#define HAVE_TDM 1
#else
#define HAVE_TDM 0
#endif

// ---------------------------- helpers --------------------------------------

__device__ __forceinline__ float sigmoidf_(float x) {
    return 1.0f / (1.0f + __expf(-x));
}
__device__ __forceinline__ float bf16lo(unsigned u) { return __uint_as_float(u << 16); }
__device__ __forceinline__ float bf16hi(unsigned u) { return __uint_as_float(u & 0xffff0000u); }

// Global arrive/release barrier for the persistent recurrence kernel.
__device__ __forceinline__ void grid_sync(unsigned* bar, unsigned nblk) {
    __threadfence();
    __syncthreads();
    if (threadIdx.x == 0) {
        volatile unsigned* vgen = bar + 1;
        unsigned g = *vgen;
        if (atomicAdd(bar, 1u) == nblk - 1u) {
            *((volatile unsigned*)bar) = 0u;
            __threadfence();
            atomicAdd(bar + 1, 1u);
        } else {
            while (*vgen == g) { __builtin_amdgcn_s_sleep(2); }
        }
        __threadfence();
    }
    __syncthreads();
}

#if HAVE_TDM
typedef __attribute__((ext_vector_type(4))) unsigned tdm_u32x4;
typedef __attribute__((ext_vector_type(8))) int      tdm_i32x8;
typedef __attribute__((ext_vector_type(4))) int      tdm_i32x4;

// Issue one TDM load of a 3D tile (d0 x d1 x d2 elements, 2-byte elements):
//   X: d0 contiguous elements, Y: d1 lines with stride s0 (elements),
//   Z: d2 planes with stride s1 (elements). Streams contiguously into LDS
//   at byte offset lds_off. D# layout per CDNA5 ISA 8.3-8.6.
__device__ __forceinline__ void tdm_load_tile_b16(unsigned lds_off,
                                                  unsigned long long gaddr,
                                                  unsigned d0, unsigned d1, unsigned d2,
                                                  unsigned long long s0,
                                                  unsigned long long s1) {
    tdm_u32x4 g0;
    g0[0] = 1u;                                              // count=1, user D#
    g0[1] = lds_off;                                         // lds_addr
    g0[2] = (unsigned)(gaddr & 0xffffffffu);                 // global_addr lo
    g0[3] = (unsigned)((gaddr >> 32) & 0x01ffffffu)          // global_addr hi
          | (2u << 30);                                      // type=2 (image)
    tdm_i32x8 g1;
    g1[0] = (int)(1u << 16);                                 // data_size=1 -> 2B
    g1[1] = (int)((d0 & 0xffffu) << 16);                     // tensor_dim0[15:0]
    g1[2] = (int)(((d0 >> 16) & 0xffffu) | ((d1 & 0xffffu) << 16)); // td0 hi | td1 lo
    g1[3] = (int)(((d1 >> 16) & 0xffffu) | ((d0 & 0xffffu) << 16)); // td1 hi | tile_dim0
    g1[4] = (int)((d1 & 0xffffu) | ((d2 & 0xffffu) << 16));  // tile_dim1 | tile_dim2
    g1[5] = (int)(s0 & 0xffffffffu);                         // dim0_stride[31:0]
    g1[6] = (int)(((s0 >> 32) & 0xffffu) | ((unsigned)(s1 & 0xffffu) << 16));
    g1[7] = (int)((s1 >> 16) & 0xffffffffu);                 // dim1_stride[47:16]
    tdm_i32x4 g2;
    g2[0] = (int)d2;                                         // tensor_dim2
    g2[1] = 0; g2[2] = 0; g2[3] = 0;
    tdm_i32x4 g3; g3[0] = 0; g3[1] = 0; g3[2] = 0; g3[3] = 0;
#if defined(__clang_major__) && (__clang_major__ >= 23)
    tdm_i32x8 g4; g4[0]=0; g4[1]=0; g4[2]=0; g4[3]=0; g4[4]=0; g4[5]=0; g4[6]=0; g4[7]=0;
    __builtin_amdgcn_tensor_load_to_lds(g0, g1, g2, g3, g4, 0);
#else
    __builtin_amdgcn_tensor_load_to_lds(g0, g1, g2, g3, 0);
#endif
}
#endif // HAVE_TDM

// ---------------------------- (1) prep -------------------------------------

__global__ void prep_kernel(const float* __restrict__ bih,
                            const float* __restrict__ bhh,
                            float* __restrict__ bias,
                            float* __restrict__ hbuf,
                            float* __restrict__ cfin,
                            unsigned* __restrict__ bar) {
    int i = blockIdx.x * 256 + threadIdx.x;               // grid: 128 blocks
    if (i < L_ * G4H) bias[i] = bih[i] + bhh[i];
    if (i < 2 * H_)   hbuf[i] = 0.0f;
    if (i < H_)       cfin[i] = 0.0f;
    if (i < 2)        bar[i]  = 0u;
}

// ---------------------------- (2) f32 -> bf16 -------------------------------

__global__ void cvt_bf16_kernel(const float* __restrict__ src,
                                unsigned short* __restrict__ dst, size_t n) {
    size_t i = (size_t)blockIdx.x * blockDim.x + threadIdx.x;
    if (i < n) {
        unsigned u = __float_as_uint(src[i]);
        unsigned r = (u + 0x7fffu + ((u >> 16) & 1u)) >> 16;   // RNE
        dst[i] = (unsigned short)r;
    }
}

// ---------------------------- (3) WMMA x-projection -------------------------
// Gx[l][t][r] = sum_k Wih[l][r][k] * x[t][k] + bias[l][r]
// Per-wave: 16 rows x 64 cols (4 tiles, A-frag reuse). Block = 8 waves in M.
// K-loop register double-buffered so loads for iter k+1 overlap WMMAs of k.

union FragU { uint4 q[2]; bf16x16 v; };

__global__ void xproj_wmma_kernel(const unsigned short* __restrict__ wih,  // [L][4H][IN] bf16
                                  const unsigned short* __restrict__ xbf,  // [T][IN] bf16
                                  const float* __restrict__ bias,          // [L][4H]
                                  float* __restrict__ gx) {                // [L][T][4H]
    const int l    = blockIdx.z;
    const int wave = threadIdx.x >> 5;
    const int lane = threadIdx.x & 31;
    const int half = lane >> 4;
    const int l15  = lane & 15;
    const int r0   = blockIdx.x * 128 + wave * 16;
    const int t0   = blockIdx.y * 64;

    const unsigned short* arow = wih + ((size_t)l * G4H + (r0 + l15)) * IN_;
    const unsigned short* bcol[4];
#pragma unroll
    for (int n = 0; n < 4; ++n)
        bcol[n] = xbf + (size_t)(t0 + n * 16 + l15) * IN_;

    FragU a[2], b[2][4];
    auto loadA = [&](FragU& f, int k) {
        f.q[0] = *(const uint4*)(arow + k + half * 8);
        f.q[1] = *(const uint4*)(arow + k + half * 8 + 16);
    };
    auto loadB = [&](FragU (&f)[4], int k) {
#pragma unroll
        for (int n = 0; n < 4; ++n) {
            const unsigned short* p = bcol[n] + k + half * 8;
            f[n].q[0] = *(const uint4*)(p);
            f[n].q[1] = *(const uint4*)(p + 16);
        }
    };

    loadA(a[0], 0);
    loadB(b[0], 0);

    f32x8 acc[4] = {f32x8{}, f32x8{}, f32x8{}, f32x8{}};
    for (int k = 0; k < IN_; k += 32) {
        const int cur = (k >> 5) & 1;
        const int nxt = cur ^ 1;
        if (k + 32 < IN_) {            // prefetch next K-iter into other regs
            loadA(a[nxt], k + 32);
            loadB(b[nxt], k + 32);
        }
#pragma unroll
        for (int n = 0; n < 4; ++n)
            acc[n] = __builtin_amdgcn_wmma_f32_16x16x32_bf16(
                false, a[cur].v, false, b[cur][n].v, (short)0, acc[n], false, false);
    }

    const int rbase = r0 + half * 8;                 // 8 contiguous M per lane
    const float* bptr = bias + (size_t)l * G4H + rbase;
#pragma unroll
    for (int n = 0; n < 4; ++n) {
        float* dst = gx + ((size_t)l * T_ + (t0 + n * 16 + l15)) * G4H + rbase;
#pragma unroll
        for (int j = 0; j < 8; ++j) dst[j] = acc[n][j] + bptr[j];
    }
}

// ---------------------------- (4) persistent recurrence ---------------------
// 64 blocks x 256 threads. Block g owns h/c elements j in [g*16, g*16+16) and
// gate rows {j, H+j, 2H+j, 3H+j}. Weight tile for step s+1 (64 rows x 2KB =
// 128KB) is TDM-loaded into the other LDS buffer while step s computes.
// h is double-buffered in global memory; c slice never leaves the block.

__global__ void lstm_recur_kernel(const unsigned short* __restrict__ whh,  // [L][4H][H] bf16
                                  const float* __restrict__ gx,            // [L][T][4H]
                                  float* __restrict__ hbuf,                // [2][H]
                                  float* __restrict__ cfin,                // [H]
                                  unsigned* __restrict__ bar) {
    extern __shared__ char smem[];
    unsigned short* wbuf0 = (unsigned short*)(smem);
    unsigned short* wbuf1 = (unsigned short*)(smem + WBYTES);
    float* hlds = (float*)(smem + 2 * WBYTES);
    float* glds = hlds + H_;
    float* clds = glds + 64;

    const int tid   = threadIdx.x;
    const int wave  = tid >> 5;
    const int lane  = tid & 31;
    const int jbase = blockIdx.x * 16;

    if (tid < 16) clds[tid] = 0.0f;     // c0 = 0
    __syncthreads();

#if HAVE_TDM
    if (wave == 0) {                    // preload step-0 weight tile (layer 0)
        tdm_load_tile_b16((unsigned)(size_t)wbuf0,
                          (unsigned long long)(size_t)(whh + (size_t)jbase * H_),
                          H_, 16, 4,
                          (unsigned long long)H_,               // row stride
                          (unsigned long long)H_ * H_);         // gate stride
    }
#endif

    int step = 0;
    for (int t = 0; t < T_; ++t) {
        for (int l = 0; l < L_; ++l) {
            const float* hread  = hbuf + (step & 1) * H_;
            float*       hwrite = hbuf + ((step & 1) ^ 1) * H_;

            for (int i = tid; i < H_; i += 256) hlds[i] = hread[i];

#if HAVE_TDM
            if (wave == 0) {
                if (step + 1 < NSTEP) {        // stage next step's weights
                    const int lnx = (l == L_ - 1) ? 0 : l + 1;
                    unsigned short* wnxt = ((step + 1) & 1) ? wbuf1 : wbuf0;
                    tdm_load_tile_b16((unsigned)(size_t)wnxt,
                        (unsigned long long)(size_t)(whh + ((size_t)lnx * G4H + jbase) * H_),
                        H_, 16, 4,
                        (unsigned long long)H_,
                        (unsigned long long)H_ * H_);
                    __builtin_amdgcn_s_wait_tensorcnt(1);   // current tile ready
                } else {
                    __builtin_amdgcn_s_wait_tensorcnt(0);
                }
            }
#endif
            __syncthreads();

            const float* gxrow = gx + ((size_t)l * T_ + t) * G4H;
#if HAVE_TDM
            const unsigned* wcur32 =
                (const unsigned*)(((step & 1) ? wbuf1 : wbuf0));
#endif
#pragma unroll
            for (int rr = 0; rr < 8; ++rr) {
                const int lr   = wave * 8 + rr;        // 0..63
                const int gate = lr >> 4;
                const int jl   = lr & 15;
                const int row  = gate * H_ + jbase + jl;

                float acc = 0.0f;
#if HAVE_TDM
                const unsigned* w32 = wcur32 + (size_t)lr * (H_ / 2);
#pragma unroll
                for (int i = 0; i < 16; ++i) {
                    const int k = i * 64 + lane * 2;
                    unsigned u = w32[i * 32 + lane];   // LDS, conflict-free
                    acc += bf16lo(u) * hlds[k] + bf16hi(u) * hlds[k + 1];
                }
#else
                const unsigned short* wrow = whh + ((size_t)l * G4H + row) * H_;
                __builtin_prefetch(wrow + H_, 0, 1);
#pragma unroll
                for (int i = 0; i < 16; ++i) {
                    const int k = i * 64 + lane * 2;
                    unsigned u = *(const unsigned*)(wrow + k);
                    acc += bf16lo(u) * hlds[k] + bf16hi(u) * hlds[k + 1];
                }
#endif
#pragma unroll
                for (int off = 16; off > 0; off >>= 1)
                    acc += __shfl_xor(acc, off);
                if (lane == 0) glds[lr] = acc + gxrow[row];
            }
            __syncthreads();

            if (tid < 16) {
                float iv = sigmoidf_(glds[tid]);
                float fv = sigmoidf_(glds[16 + tid]);
                float gv = tanhf(glds[32 + tid]);
                float ov = sigmoidf_(glds[48 + tid]);
                float cv = fv * clds[tid] + iv * gv;
                clds[tid] = cv;
                hwrite[jbase + tid] = ov * tanhf(cv);
            }

            grid_sync(bar, NBLK_R);
            ++step;
        }
    }

    if (tid < 16) cfin[jbase + tid] = clds[tid];   // final cell state
}

// ---------------------------- (5) output head -------------------------------

__global__ void head_kernel(const float* __restrict__ wout,   // [OUT][H]
                            const float* __restrict__ bout,   // [OUT]
                            const float* __restrict__ cvec,   // [H]
                            float* __restrict__ out) {        // [OUT]
    const int row  = blockIdx.x * 8 + (threadIdx.x >> 5);     // 128 blocks
    const int lane = threadIdx.x & 31;
    const float* wrow = wout + (size_t)row * H_;
    float acc = 0.0f;
    for (int k = lane; k < H_; k += 32) acc += wrow[k] * cvec[k];
#pragma unroll
    for (int off = 16; off > 0; off >>= 1) acc += __shfl_xor(acc, off);
    if (lane == 0) out[row] = acc + bout[row];
}

// ---------------------------- launch ----------------------------------------

extern "C" void kernel_launch(void* const* d_in, const int* in_sizes, int n_in,
                              void* d_out, int out_size, void* d_ws, size_t ws_size,
                              hipStream_t stream) {
    const float* x    = (const float*)d_in[0];   // [1,T,IN]
    const float* Wih  = (const float*)d_in[1];   // [L,4H,IN]
    const float* Whh  = (const float*)d_in[2];   // [L,4H,H]
    const float* bih  = (const float*)d_in[3];   // [L,4H]
    const float* bhh  = (const float*)d_in[4];   // [L,4H]
    const float* Wout = (const float*)d_in[5];   // [OUT,H]
    const float* bout = (const float*)d_in[6];   // [OUT]
    float* out = (float*)d_out;                  // [OUT]

    char* ws = (char*)d_ws;
    size_t off = 0;
    auto take = [&](size_t bytes) {
        char* p = ws + off;
        off = (off + bytes + 255) & ~(size_t)255;
        return p;
    };
    unsigned short* xbf   = (unsigned short*)take((size_t)T_ * IN_ * 2);        //   4 MB
    unsigned short* wihbf = (unsigned short*)take((size_t)L_ * G4H * IN_ * 2);  //  64 MB
    unsigned short* whhbf = (unsigned short*)take((size_t)L_ * G4H * H_ * 2);   //  64 MB
    float*    bias = (float*)take((size_t)L_ * G4H * 4);                        // 128 KB
    float*    hbuf = (float*)take((size_t)2 * H_ * 4);
    float*    cfin = (float*)take((size_t)H_ * 4);
    unsigned* bar  = (unsigned*)take(256);
    float*    gxp  = (float*)take((size_t)L_ * T_ * G4H * 4);                   // 256 MB

    prep_kernel<<<128, 256, 0, stream>>>(bih, bhh, bias, hbuf, cfin, bar);

    {
        size_t n = (size_t)T_ * IN_;
        cvt_bf16_kernel<<<(unsigned)((n + 255) / 256), 256, 0, stream>>>(x, xbf, n);
    }
    {
        size_t n = (size_t)L_ * G4H * IN_;
        cvt_bf16_kernel<<<(unsigned)((n + 255) / 256), 256, 0, stream>>>(Wih, wihbf, n);
    }
    {
        size_t n = (size_t)L_ * G4H * H_;
        cvt_bf16_kernel<<<(unsigned)((n + 255) / 256), 256, 0, stream>>>(Whh, whhbf, n);
    }

    xproj_wmma_kernel<<<dim3(32, 32, L_), 256, 0, stream>>>(wihbf, xbf, bias, gxp);

    lstm_recur_kernel<<<NBLK_R, 256, DYN_LDS, stream>>>(whhbf, gxp, hbuf, cfin, bar);

    head_kernel<<<128, 256, 0, stream>>>(Wout, bout, cfin, out);
}
